// TrajectoryDecoder_32950989095615
// MI455X (gfx1250) — matrix operand order repaired and checked by reading, not compiled, not run on hardware
//
#include <hip/hip_runtime.h>

// ---------------------------------------------------------------------------
// TrajectoryDecoder: 2-layer LSTM (H=128) rolled out T=30 steps, B=65536.
// Persistent per-wave 16-row batch tile; h-state bf16 in LDS, c-state fp32 in
// LDS; weights pre-packed to bf16 WMMA B-fragment layout in workspace
// (L2/WGP$-resident, 384 KB, re-read per step). An opaque laundered zero
// offset per hc-tile breaks loop-invariance (prevents LICM hoist + scratch
// spill) while preserving address-space inference (global_load, not flat).
// Matrix ops: v_wmma_f32_16x16x32_bf16 (CDNA5, wave32).
// ---------------------------------------------------------------------------

#define BATCH   65536
#define HDIM    128
#define TSTEPS  30
#define WPB     2          // waves per block
#define RPW     16         // batch rows per wave (WMMA M)

typedef __attribute__((ext_vector_type(16))) __bf16 v16bf;
typedef __attribute__((ext_vector_type(8)))  __bf16 v8bf;
typedef __attribute__((ext_vector_type(8)))  float  v8f;

#define WMMA_BF16(A, B, C) \
  __builtin_amdgcn_wmma_f32_16x16x32_bf16(false, (A), false, (B), (short)0, (C), false, false)

// Opaque zero: breaks loop-invariance of loads whose address depends on it,
// without destroying pointer address-space inference (keeps global/ds loads).
__device__ __forceinline__ int lzero() {
  int v = 0;
  asm volatile("" : "+s"(v));
  return v;
}

__device__ __forceinline__ unsigned short f2bf(float f) {
  union { float f; unsigned u; } v; v.f = f;
  unsigned u = v.u;
  unsigned r = u + 0x7FFFu + ((u >> 16) & 1u);   // round-to-nearest-even
  return (unsigned short)(r >> 16);
}

__device__ __forceinline__ float sigm(float x) {
  return 1.0f / (1.0f + __expf(-x));
}
__device__ __forceinline__ float tanh_fast(float x) {
  return 2.0f / (1.0f + __expf(-2.0f * x)) - 1.0f;
}

// A-fragment (16x32 bf16) from a row-major 16x128 bf16 tile in LDS.
// A layout (matches packed B): lanes 0-15 (M=row) hold K 0..15,
// lanes 16-31 hold K 16..31 of the same rows.
__device__ __forceinline__ v16bf load_afrag(const unsigned short* tile,
                                            int nlo, int hi, int ko) {
  return *(const v16bf*)(tile + nlo * HDIM + ko + hi * 16);
}

// ---------------------------------------------------------------------------
// Weight packer: fp32 -> bf16 in WMMA B-fragment order.
// B-fragment (32x16, KxN): lane l (0..15) holds N=l, K contiguous kbase..+15;
// lane l (16..31) holds N=l-16, K kbase+16..+31 (this code's contiguous k order, same as A).
// Layer0 B-buffer: W_hh0 (512x128), frag = nt*4+kk.
// Layer1 B-buffer: [W_ih1; W_hh1] stacked (K=256),  frag = nt*8+kk.
// ---------------------------------------------------------------------------
__global__ void pack_weights(const float* __restrict__ Whh0,
                             const float* __restrict__ Wih1,
                             const float* __restrict__ Whh1,
                             unsigned short* __restrict__ pW0,
                             unsigned short* __restrict__ pW1) {
  int idx = blockIdx.x * 256 + threadIdx.x;        // 0 .. 12287
  if (idx < 4096) {                                // layer 0: 32 nt * 4 kk * 32 lanes
    int lane = idx & 31, f = idx >> 5;
    int nt = f >> 2, kk = f & 3;
    int n = nt * 16 + (lane & 15);
    int kbase = kk * 32 + (lane >> 4) * 16;
    unsigned short* dst = pW0 + idx * 16;
    #pragma unroll
    for (int j = 0; j < 16; ++j)
      dst[j] = f2bf(Whh0[n * HDIM + kbase + j]);
  } else {                                         // layer 1: 32 nt * 8 kk * 32 lanes
    int t = idx - 4096;
    int lane = t & 31, f = t >> 5;
    int nt = f >> 3, kk = f & 7;
    int n = nt * 16 + (lane & 15);
    int kbase = kk * 32 + (lane >> 4) * 16;
    unsigned short* dst = pW1 + t * 16;
    #pragma unroll
    for (int j = 0; j < 16; ++j) {
      int k = kbase + j;
      float w = (k < HDIM) ? Wih1[n * HDIM + k] : Whh1[n * HDIM + (k - HDIM)];
      dst[j] = f2bf(w);
    }
  }
}

// ---------------------------------------------------------------------------
// Main persistent-state rollout kernel.
// ---------------------------------------------------------------------------
__launch_bounds__(WPB * 32)
__global__ void traj_rollout(const float* __restrict__ h_n,
                             const float* __restrict__ c_n,
                             const float* __restrict__ last_pos,
                             const float* __restrict__ last_vel,
                             const float* __restrict__ W_ih0,
                             const float* __restrict__ b_ih0,
                             const float* __restrict__ b_hh0,
                             const float* __restrict__ b_ih1,
                             const float* __restrict__ b_hh1,
                             const float* __restrict__ W_out,
                             const float* __restrict__ b_out,
                             const unsigned short* __restrict__ pW0,
                             const unsigned short* __restrict__ pW1,
                             float* __restrict__ out) {
  __shared__ __align__(32) unsigned short sh0[WPB][RPW * HDIM];   // h0 (bf16)
  __shared__ __align__(32) unsigned short sh1[WPB][RPW * HDIM];   // h1 (bf16)
  __shared__ __align__(32) unsigned short stage[WPB][RPW * HDIM]; // h staging
  __shared__ float sc0[WPB][RPW * HDIM];                          // c0 (fp32)
  __shared__ float sc1[WPB][RPW * HDIM];                          // c1 (fp32)
  __shared__ float sbias0[512];
  __shared__ float sbias1[512];
  __shared__ float sx[WPB][RPW][2];
  __shared__ float spos[WPB][RPW][2];
  __shared__ float svel[WPB][RPW][2];
  __shared__ float sres[WPB][RPW][2];

  const int tid  = threadIdx.x;
  const int wave = tid >> 5;
  const int lane = tid & 31;
  const int nlo  = lane & 15;           // N-column within tile / M-row for A
  const int hi   = lane >> 4;           // lane half
  const int row0 = blockIdx.x * (WPB * RPW) + wave * RPW;

  // Block-cooperative preload of biases.
  for (int i = tid; i < 512; i += WPB * 32) {
    sbias0[i] = b_ih0[i] + b_hh0[i];
    sbias1[i] = b_ih1[i] + b_hh1[i];
  }
  // Per-wave h/c state -> LDS tiles (h as bf16, c as fp32).
  for (int e = lane; e < RPW * HDIM; e += 32) {
    int m = e >> 7, k = e & (HDIM - 1);
    size_t o = (size_t)(row0 + m) * HDIM + k;
    sh0[wave][e] = f2bf(h_n[o]);
    sh1[wave][e] = f2bf(h_n[(size_t)BATCH * HDIM + o]);
    sc0[wave][e] = c_n[o];
    sc1[wave][e] = c_n[(size_t)BATCH * HDIM + o];
  }
  if (lane < RPW) {
    size_t o = (size_t)(row0 + lane) * 2;
    spos[wave][lane][0] = last_pos[o];     spos[wave][lane][1] = last_pos[o + 1];
    svel[wave][lane][0] = last_vel[o];     svel[wave][lane][1] = last_vel[o + 1];
    sx[wave][lane][0]   = spos[wave][lane][0];
    sx[wave][lane][1]   = spos[wave][lane][1];
  }

  // Output head weights per lane column.
  float wout0[8], wout1[8];
  #pragma unroll
  for (int hc = 0; hc < 8; ++hc) {
    int col = hc * 16 + nlo;
    wout0[hc] = W_out[col];
    wout1[hc] = W_out[HDIM + col];
  }
  const float bo0 = b_out[0], bo1 = b_out[1];

  __syncthreads();

  for (int t = 0; t < TSTEPS; ++t) {
    if (lane < RPW) { sres[wave][lane][0] = 0.f; sres[wave][lane][1] = 0.f; }

    float xr0[8], xr1[8];
    #pragma unroll
    for (int r = 0; r < 8; ++r) {
      int m = r + 8 * hi;
      xr0[r] = sx[wave][m][0];
      xr1[r] = sx[wave][m][1];
    }

    // ---------------- layer 0: gates = x@W_ih0^T + h0@W_hh0^T + b ----------
    #pragma unroll
    for (int hc = 0; hc < 8; ++hc) {
      const int z = lzero();                              // defeat LICM per tile
      const unsigned short* pw0h = pW0 + z;               // stays global addrspace
      const float* wihh = W_ih0 + z;
      const float* pb0h = &sbias0[0] + z;                 // stays LDS addrspace
      v8f acc[4];
      #pragma unroll
      for (int g = 0; g < 4; ++g) {
        int n = g * HDIM + hc * 16 + nlo;
        float b = pb0h[n];
        float w0 = wihh[2 * n], w1 = wihh[2 * n + 1];     // L2-cached
        #pragma unroll
        for (int r = 0; r < 8; ++r)
          acc[g][r] = b + xr0[r] * w0 + xr1[r] * w1;      // fold K=2 input GEMM
      }
      for (int kk = 0; kk < 4; ++kk) {
        v16bf a = load_afrag(&sh0[wave][0], nlo, hi, kk * 32);
        #pragma unroll
        for (int g = 0; g < 4; ++g) {
          int nt = g * 8 + hc;
          v16bf bb = *(const v16bf*)(pw0h + (((nt * 4 + kk) * 32 + lane) << 4));
          acc[g] = WMMA_BF16(a, bb, acc[g]);
        }
      }
      #pragma unroll
      for (int r = 0; r < 8; ++r) {
        int m = r + 8 * hi;
        int ci = m * HDIM + hc * 16 + nlo;
        float iv = sigm(acc[0][r]);
        float fv = sigm(acc[1][r]);
        float gv = tanh_fast(acc[2][r]);
        float ov = sigm(acc[3][r]);
        float c  = fv * sc0[wave][ci] + iv * gv;
        sc0[wave][ci] = c;
        float h  = ov * tanh_fast(c);
        stage[wave][ci] = f2bf(h);                        // h0_new -> staging
      }
    }
    // commit h0_new (stage) -> sh0 (layer0 reads done; same-wave DS in-order)
    #pragma unroll
    for (int i = 0; i < 4; ++i) {
      int e = (i * 32 + lane) * 16;
      *(v16bf*)&sh0[wave][e] = *(const v16bf*)&stage[wave][e];
    }

    // ---------------- layer 1: gates = [h0_new;h1] @ [Wih1;Whh1]^T + b -----
    float res0[8], res1[8];
    #pragma unroll
    for (int r = 0; r < 8; ++r) { res0[r] = 0.f; res1[r] = 0.f; }

    #pragma unroll
    for (int hc = 0; hc < 8; ++hc) {
      const int z = lzero();                              // defeat LICM per tile
      const unsigned short* pw1h = pW1 + z;
      const float* pb1h = &sbias1[0] + z;
      v8f acc[4];
      #pragma unroll
      for (int g = 0; g < 4; ++g) {
        float b = pb1h[g * HDIM + hc * 16 + nlo];
        #pragma unroll
        for (int r = 0; r < 8; ++r) acc[g][r] = b;
      }
      for (int kk = 0; kk < 8; ++kk) {
        const unsigned short* hsrc = (kk < 4) ? &sh0[wave][0] : &sh1[wave][0];
        v16bf a = load_afrag(hsrc, nlo, hi, (kk & 3) * 32);
        #pragma unroll
        for (int g = 0; g < 4; ++g) {
          int nt = g * 8 + hc;
          v16bf bb = *(const v16bf*)(pw1h + (((nt * 8 + kk) * 32 + lane) << 4));
          acc[g] = WMMA_BF16(a, bb, acc[g]);
        }
      }
      #pragma unroll
      for (int r = 0; r < 8; ++r) {
        int m = r + 8 * hi;
        int ci = m * HDIM + hc * 16 + nlo;
        float iv = sigm(acc[0][r]);
        float fv = sigm(acc[1][r]);
        float gv = tanh_fast(acc[2][r]);
        float ov = sigm(acc[3][r]);
        float c  = fv * sc1[wave][ci] + iv * gv;
        sc1[wave][ci] = c;
        float h  = ov * tanh_fast(c);
        stage[wave][ci] = f2bf(h);                        // h1_new -> staging
        res0[r] += h * wout0[hc];                         // output head partials
        res1[r] += h * wout1[hc];
      }
    }
    // commit h1_new -> sh1
    #pragma unroll
    for (int i = 0; i < 4; ++i) {
      int e = (i * 32 + lane) * 16;
      *(v16bf*)&sh1[wave][e] = *(const v16bf*)&stage[wave][e];
    }

    // Reduce output head across the 16 lanes of each half (ds_add_f32).
    #pragma unroll
    for (int r = 0; r < 8; ++r) {
      int m = r + 8 * hi;
      atomicAdd(&sres[wave][m][0], res0[r]);
      atomicAdd(&sres[wave][m][1], res1[r]);
    }

    // next_pos = last_pos + last_vel*(t+1) + (h1@W_out^T + b_out)
    if (lane < RPW) {
      int m = lane;
      float tv = (float)(t + 1);
      float np0 = spos[wave][m][0] + svel[wave][m][0] * tv + sres[wave][m][0] + bo0;
      float np1 = spos[wave][m][1] + svel[wave][m][1] * tv + sres[wave][m][1] + bo1;
      size_t ob = ((size_t)(row0 + m) * TSTEPS + t) * 2;
      out[ob]     = np0;
      out[ob + 1] = np1;
      sx[wave][m][0] = np0;
      sx[wave][m][1] = np1;
    }
  }
}

// ---------------------------------------------------------------------------
extern "C" void kernel_launch(void* const* d_in, const int* in_sizes, int n_in,
                              void* d_out, int out_size, void* d_ws, size_t ws_size,
                              hipStream_t stream) {
  (void)in_sizes; (void)n_in; (void)out_size; (void)ws_size;

  const float* h_n      = (const float*)d_in[0];
  const float* c_n      = (const float*)d_in[1];
  const float* last_pos = (const float*)d_in[2];
  const float* last_vel = (const float*)d_in[3];
  const float* W_ih0    = (const float*)d_in[4];
  const float* W_hh0    = (const float*)d_in[5];
  const float* b_ih0    = (const float*)d_in[6];
  const float* b_hh0    = (const float*)d_in[7];
  const float* W_ih1    = (const float*)d_in[8];
  const float* W_hh1    = (const float*)d_in[9];
  const float* b_ih1    = (const float*)d_in[10];
  const float* b_hh1    = (const float*)d_in[11];
  const float* W_out    = (const float*)d_in[12];
  const float* b_out    = (const float*)d_in[13];
  float* out            = (float*)d_out;

  unsigned short* pW0 = (unsigned short*)d_ws;          // 32*4*32*16  = 64K halves
  unsigned short* pW1 = pW0 + 32 * 4 * 32 * 16;         // 32*8*32*16  = 128K halves

  pack_weights<<<48, 256, 0, stream>>>(W_hh0, W_ih1, W_hh1, pW0, pW1);

  int blocks = BATCH / (WPB * RPW);                     // 2048
  traj_rollout<<<blocks, WPB * 32, 0, stream>>>(
      h_n, c_n, last_pos, last_vel,
      W_ih0, b_ih0, b_hh0, b_ih1, b_hh1,
      W_out, b_out, pW0, pW1, out);
}
